// BahdanauAttention_2233382994669
// MI455X (gfx1250) — matrix operand order repaired and checked by reading, not compiled, not run on hardware
//
#include <hip/hip_runtime.h>

// Problem dims (fixed by the reference setup)
#define B_  32
#define S_  2048
#define D_  1024
#define U_  1024
#define M_  (B_ * S_)   // 65536 flattened rows of the big GEMM
#define KP_ 256         // K-panel staged in LDS

typedef __attribute__((ext_vector_type(16))) __bf16 v16bf;
typedef __attribute__((ext_vector_type(8)))  __bf16 v8bf;
typedef __attribute__((ext_vector_type(8)))  float  v8f;
typedef __attribute__((ext_vector_type(4)))  float  v4f;

// Native hardware converts (v_cvt_*_bf16 path, RTNE) — no bit twiddling.
__device__ __forceinline__ __bf16 f2bf(float f) { return (__bf16)f; }
__device__ __forceinline__ float  bf2f(__bf16 b) { return (float)b; }

// ---------------------------------------------------------------------------
// W_enc [D,U] fp32 (row-major, k-major)  ->  Wt [U,D] bf16 (n-major, k contig)
// LDS-tiled 32x32 transpose so both global read and write are coalesced.
// ---------------------------------------------------------------------------
__global__ void __launch_bounds__(256)
k_transpose_w(const float* __restrict__ W, __bf16* __restrict__ Wt) {
  __shared__ float tile[32][33];
  const int kt = blockIdx.x * 32, nt = blockIdx.y * 32;
  const int tx = threadIdx.x, ty = threadIdx.y;          // block (32,8)
  #pragma unroll
  for (int i = 0; i < 32; i += 8)
    tile[ty + i][tx] = W[(size_t)(kt + ty + i) * U_ + nt + tx];
  __syncthreads();
  #pragma unroll
  for (int i = 0; i < 32; i += 8)
    Wt[(size_t)(nt + ty + i) * D_ + kt + tx] = f2bf(tile[tx][ty + i]);
}

// ---------------------------------------------------------------------------
// dec_p[b][u] = h_dec[b,:] @ W_dec[:,u] + b_dec[u]   (tiny: 67 MFLOP, fp32)
// ---------------------------------------------------------------------------
__global__ void __launch_bounds__(256)
k_decp(const float* __restrict__ hdec, const float* __restrict__ Wdec,
       const float* __restrict__ bdec, float* __restrict__ decp) {
  const int idx = blockIdx.x * 256 + threadIdx.x;        // B_*U_ threads
  const int b = idx >> 10, u = idx & (U_ - 1);
  const float* h = hdec + b * D_;
  float acc = bdec[u];
  for (int k = 0; k < D_; ++k) acc += h[k] * Wdec[k * U_ + u];
  decp[idx] = acc;
}

// ---------------------------------------------------------------------------
// Main fused kernel (v3): 32 rows x 1024 cols per workgroup.
// - Wave w owns cols [w*128, w*128+128) (8 n-tiles) and BOTH 16-row M-subtiles
//   => every B fragment feeds 2 wmmas (halved B traffic per FLOP).
// - A is staged per 256-wide K-panel in LDS as bf16, converted ONCE
//   cooperatively (128 native cvts/thread total) instead of per-wave in the
//   K-loop => K-loop has zero conversion VALU: 4 ds_load_b128 + B loads + wmma.
// - +8-half row pad => row stride 132 dwords == 4 mod 64 banks: conflict-free.
// ---------------------------------------------------------------------------
__global__ void __launch_bounds__(256)
k_encp_score(const float* __restrict__ h_enc, const __bf16* __restrict__ Wt,
             const float* __restrict__ b_enc, const float* __restrict__ decp,
             const float* __restrict__ Wcom, const float* __restrict__ bcom,
             __bf16* __restrict__ encp, float* __restrict__ score) {
  __shared__ __align__(16) __bf16 ldsA[32][KP_ + 8];     // ~17 KB
  __shared__ float racc[32][8];                          // 1 KB

  const int tid  = threadIdx.x;
  const int wave = tid >> 5;                             // wave32!
  const int lane = tid & 31;
  const int m0   = blockIdx.x * 32;                      // global row-tile base
  const int b    = m0 >> 11;                             // m0 / S_ (tiles never span b)

  const int half  = lane >> 4;                           // A/B half-lane K split
  const int l16   = lane & 15;
  const int nbase = wave * 128;

  v8f zero = {};
  v8f acc[2][8];
  #pragma unroll
  for (int ms = 0; ms < 2; ++ms)
    #pragma unroll
    for (int t = 0; t < 8; ++t) acc[ms][t] = zero;

  // A layout used here (contiguous k, same as B): lane holds row m (=l16 / 16+l16); elem j (0..15) <->
  // k = k0 + 16*half + j (matches B).
  const __bf16* arow0 = &ldsA[l16][16 * half];
  const __bf16* arow1 = &ldsA[16 + l16][16 * half];

  for (int kp = 0; kp < D_; kp += KP_) {
    __syncthreads();                                     // LDS reuse guard
    // Stage panel: 32 rows x 256 k, fp32 -> bf16, fully coalesced v4 loads.
    #pragma unroll
    for (int it = 0; it < (32 * KP_) / (256 * 4); ++it) { // 8 iters
      const int e  = (tid + it * 256) * 4;
      const int m  = e >> 8;                              // / KP_
      const int kk = e & (KP_ - 1);
      v4f x = *(const v4f*)(h_enc + (size_t)(m0 + m) * D_ + kp + kk);
      ldsA[m][kk + 0] = f2bf(x[0]);
      ldsA[m][kk + 1] = f2bf(x[1]);
      ldsA[m][kk + 2] = f2bf(x[2]);
      ldsA[m][kk + 3] = f2bf(x[3]);
    }
    __syncthreads();

    for (int kk0 = 0; kk0 < KP_; kk0 += 32) {
      v8bf lo0 = *(const v8bf*)(arow0 + kk0);
      v8bf hi0 = *(const v8bf*)(arow0 + kk0 + 8);
      v8bf lo1 = *(const v8bf*)(arow1 + kk0);
      v8bf hi1 = *(const v8bf*)(arow1 + kk0 + 8);
      v16bf af0 = __builtin_shufflevector(lo0, hi0,
          0,1,2,3,4,5,6,7,8,9,10,11,12,13,14,15);
      v16bf af1 = __builtin_shufflevector(lo1, hi1,
          0,1,2,3,4,5,6,7,8,9,10,11,12,13,14,15);
      const int k0 = kp + kk0;
      #pragma unroll
      for (int t = 0; t < 8; ++t) {
        // B column-major: lane holds col n = l16; elem j <-> k = k0+16*half+j
        const __bf16* bp = Wt + (size_t)(nbase + t * 16 + l16) * D_ + k0 + 16 * half;
        v16bf bfrag = *(const v16bf*)bp;
        acc[0][t] = __builtin_amdgcn_wmma_f32_16x16x32_bf16(
            false, af0, false, bfrag, (short)0, acc[0][t], false, false);
        acc[1][t] = __builtin_amdgcn_wmma_f32_16x16x32_bf16(
            false, af1, false, bfrag, (short)0, acc[1][t], false, false);
      }
    }
  }

  // Epilogue: C/D layout elem r <-> row m = msub*16 + r + 8*half, col n = l16.
  float part[2][8];
  #pragma unroll
  for (int ms = 0; ms < 2; ++ms)
    #pragma unroll
    for (int r = 0; r < 8; ++r) part[ms][r] = 0.f;

  #pragma unroll
  for (int t = 0; t < 8; ++t) {
    const int n  = nbase + t * 16 + l16;
    const float be = b_enc[n];
    const float dp = decp[b * U_ + n];
    const float wc = Wcom[n];
    #pragma unroll
    for (int ms = 0; ms < 2; ++ms) {
      #pragma unroll
      for (int r = 0; r < 8; ++r) {
        const int m = ms * 16 + r + 8 * half;
        const float e = acc[ms][t][r] + be;              // enc_p value
        encp[(size_t)(m0 + m) * U_ + n] = f2bf(e);
        part[ms][r] += tanhf(e + dp) * wc;               // fused score partial
      }
    }
  }

  // Butterfly reduce over the 16 lanes of each half (masks < 16 stay in-half).
  #pragma unroll
  for (int ms = 0; ms < 2; ++ms)
    #pragma unroll
    for (int r = 0; r < 8; ++r) {
      float v = part[ms][r];
      v += __shfl_xor(v, 1, 32);
      v += __shfl_xor(v, 2, 32);
      v += __shfl_xor(v, 4, 32);
      v += __shfl_xor(v, 8, 32);
      part[ms][r] = v;
    }
  if (l16 == 0) {                                        // lanes 0 and 16
    #pragma unroll
    for (int ms = 0; ms < 2; ++ms)
      #pragma unroll
      for (int r = 0; r < 8; ++r)
        racc[ms * 16 + r + 8 * half][wave] = part[ms][r];
  }
  __syncthreads();

  // Deterministic fixed-order cross-wave reduction: 32 threads, 1 row each.
  if (tid < 32) {
    float s = bcom[0];
    #pragma unroll
    for (int w = 0; w < 8; ++w) s += racc[tid][w];
    score[m0 + tid] = s;
  }
}

// ---------------------------------------------------------------------------
// Softmax over seq axis per batch; writes attn straight into its d_out slice.
// ---------------------------------------------------------------------------
__global__ void __launch_bounds__(256)
k_softmax(const float* __restrict__ score, float* __restrict__ attn) {
  __shared__ float red[256];
  const int b = blockIdx.x, tid = threadIdx.x;
  const float* sc = score + b * S_;
  float* ao = attn + b * S_;

  float mx = -3.4e38f;
  for (int s = tid; s < S_; s += 256) mx = fmaxf(mx, sc[s]);
  red[tid] = mx; __syncthreads();
  for (int o = 128; o > 0; o >>= 1) {
    if (tid < o) red[tid] = fmaxf(red[tid], red[tid + o]);
    __syncthreads();
  }
  mx = red[0]; __syncthreads();

  float sum = 0.f;
  for (int s = tid; s < S_; s += 256) {
    const float e = expf(sc[s] - mx);
    ao[s] = e; sum += e;
  }
  red[tid] = sum; __syncthreads();
  for (int o = 128; o > 0; o >>= 1) {
    if (tid < o) red[tid] += red[tid + o];
    __syncthreads();
  }
  const float inv = 1.f / red[0];
  for (int s = tid; s < S_; s += 256) ao[s] *= inv;
}

// ---------------------------------------------------------------------------
// context partials: split S into 8 chunks for parallelism; deterministic.
// ---------------------------------------------------------------------------
__global__ void __launch_bounds__(256)
k_ctx_partial(const __bf16* __restrict__ encp, const float* __restrict__ attn,
              float* __restrict__ partial) {
  const int b = blockIdx.x, uc = blockIdx.y, scnk = blockIdx.z;
  const int tid = threadIdx.x;
  const int u = uc * 256 + tid;
  const float* ab = attn + b * S_;
  const __bf16* eb = encp + (size_t)b * S_ * U_ + u;
  const int s0 = scnk * 256;
  float acc = 0.f;
  for (int s = s0; s < s0 + 256; ++s)
    acc += ab[s] * bf2f(eb[(size_t)s * U_]);
  partial[(size_t)((b * 4 + uc) * 256 + tid) * 8 + scnk] = acc;   // = (b*U_+u)*8+chunk
}

__global__ void __launch_bounds__(256)
k_ctx_final(const float* __restrict__ partial, float* __restrict__ ctx) {
  const int idx = blockIdx.x * 256 + threadIdx.x;        // B_*U_
  const float* p = partial + (size_t)idx * 8;
  float s = 0.f;
  #pragma unroll
  for (int i = 0; i < 8; ++i) s += p[i];
  ctx[idx] = s;
}

// ---------------------------------------------------------------------------
extern "C" void kernel_launch(void* const* d_in, const int* in_sizes, int n_in,
                              void* d_out, int out_size, void* d_ws, size_t ws_size,
                              hipStream_t stream) {
  const float* h_enc = (const float*)d_in[0];
  const float* h_dec = (const float*)d_in[1];
  const float* W_enc = (const float*)d_in[2];
  const float* b_enc = (const float*)d_in[3];
  const float* W_dec = (const float*)d_in[4];
  const float* b_dec = (const float*)d_in[5];
  const float* W_com = (const float*)d_in[6];
  const float* b_com = (const float*)d_in[7];

  float* ctx_out  = (float*)d_out;                 // [B,U]
  float* attn_out = (float*)d_out + B_ * U_;       // [B,S,1]

  // Workspace layout (all offsets 256B-aligned)
  char* ws = (char*)d_ws;
  __bf16* Wt      = (__bf16*)ws;                                       //   2 MB
  float*  decp    = (float*)(ws + (size_t)2 * 1024 * 1024);            // 128 KB
  float*  score   = (float*)(ws + (size_t)2 * 1024 * 1024 + 131072);   // 256 KB
  float*  partial = (float*)(ws + (size_t)2 * 1024 * 1024 + 131072 + 262144);            // 4 MB
  __bf16* encp    = (__bf16*)(ws + (size_t)2 * 1024 * 1024 + 131072 + 262144
                                 + (size_t)4 * 1024 * 1024);           // 134 MB bf16

  k_transpose_w<<<dim3(D_ / 32, U_ / 32), dim3(32, 8), 0, stream>>>(W_enc, Wt);
  k_decp<<<(B_ * U_) / 256, 256, 0, stream>>>(h_dec, W_dec, b_dec, decp);
  k_encp_score<<<M_ / 32, 256, 0, stream>>>(h_enc, Wt, b_enc, decp,
                                            W_com, b_com, encp, score);
  k_softmax<<<B_, 256, 0, stream>>>(score, attn_out);
  k_ctx_partial<<<dim3(B_, U_ / 256, S_ / 256), 256, 0, stream>>>(encp, attn_out, partial);
  k_ctx_final<<<(B_ * U_) / 256, 256, 0, stream>>>(partial, ctx_out);
}
